// CausalSelfAttention_59141699666117
// MI455X (gfx1250) — compile-verified, matrix operand order checked by reading, not run on hardware
//
#include <hip/hip_runtime.h>
#include <hip/hip_bf16.h>

// ---------------------------------------------------------------------------
// Causal self-attention for MI455X (gfx1250, wave32, WMMA + TDM).
//   B=2, T=2048, C=1024, H=16, D=64
// Pipeline: f32->f16 convert (+weight transpose), TDM-fed WMMA GEMM (qkv),
//           TDM-fed flash-attention (WMMA), TDM-fed WMMA GEMM (out proj).
// ---------------------------------------------------------------------------

typedef __attribute__((ext_vector_type(16))) _Float16      v16h;
typedef __attribute__((ext_vector_type(8)))  _Float16      v8h;
typedef __attribute__((ext_vector_type(8)))  float         v8f;
typedef __attribute__((ext_vector_type(4)))  unsigned int  u32x4;
typedef __attribute__((ext_vector_type(8)))  int           i32x8;
typedef __attribute__((ext_vector_type(4)))  int           i32x4;

#define CB   1024      // C
#define TT_  2048      // T
#define HH_  16        // heads
#define DD_  64        // head dim
#define BB_  2         // batch

static __device__ __forceinline__ v8f wmma_f16(const v16h& a, const v16h& b, const v8f& c) {
    return __builtin_amdgcn_wmma_f32_16x16x32_f16(false, a, false, b, (short)0, c, false, false);
}

// ---- Tensor Data Mover: 2D f16 tile global->LDS with padded rows ----------
// tile_w elems per row, tile_h rows, row_stride elems in global,
// pad codes per ISA D# group1: pad_interval 0=2dw,1=4dw,...; pad_amount 0=1dw,...
static __device__ __forceinline__ void tdm_load_2d(const _Float16* gsrc, void* lds_dst,
                                                   int tile_w, int tile_h, int row_stride,
                                                   int pad_interval, int pad_amount) {
    unsigned long long ga = (unsigned long long)(uintptr_t)gsrc;
    u32x4 g0;
    g0[0] = 1u;                                         // count=1, user descriptor
    g0[1] = (unsigned)(uintptr_t)lds_dst;               // LDS byte offset (low 32 bits)
    g0[2] = (unsigned)ga;                               // global addr [95:64]
    g0[3] = (unsigned)((ga >> 32) & 0x01FFFFFFu) | 0x80000000u;  // addr hi + type=2
    i32x8 g1;
    g1[0] = 0x00010000                                  // data_size = 2 bytes
          | 0x00100000                                  // pad_enable
          | (pad_interval << 22) | (pad_amount << 25);
    g1[1] = (tile_w & 0xFFFF) << 16;                    // tensor_dim0 (== tile, in-bounds)
    g1[2] = ((tile_w >> 16) & 0xFFFF) | ((tile_h & 0xFFFF) << 16);   // dim0 hi | dim1 lo
    g1[3] = ((tile_h >> 16) & 0xFFFF) | ((tile_w & 0xFFFF) << 16);   // dim1 hi | tile_dim0
    g1[4] = (tile_h & 0xFFFF);                          // tile_dim1 | tile_dim2=0
    g1[5] = row_stride;                                 // tensor_dim0_stride lo32
    g1[6] = 0;
    g1[7] = 0;
    i32x4 z4 = {0, 0, 0, 0};
    i32x8 z8 = {0, 0, 0, 0, 0, 0, 0, 0};
    // 6-arg form (clang-23 / therock-10.0 headers): (g0, g1, g2, g3, g4, cpol)
    __builtin_amdgcn_tensor_load_to_lds(g0, g1, z4, z4, z8, 0);
}

// --------------------------- f32 -> f16 convert ----------------------------
__global__ void cvt_f32_f16(const float* __restrict__ src, _Float16* __restrict__ dst, int n) {
    int i = (blockIdx.x * blockDim.x + threadIdx.x) * 4;
    if (i + 3 < n) {
        float4 v = *(const float4*)(src + i);
        dst[i + 0] = (_Float16)v.x;
        dst[i + 1] = (_Float16)v.y;
        dst[i + 2] = (_Float16)v.z;
        dst[i + 3] = (_Float16)v.w;
    }
}

// ---- f32[K][N] -> f16[N][K] tiled transpose-convert (weights, one-time) ---
__global__ void transpose_cvt(const float* __restrict__ src, _Float16* __restrict__ dst,
                              int K, int N) {
    __shared__ float tile[32][33];
    const int bn = blockIdx.x * 32;     // N
    const int bk = blockIdx.y * 32;     // K
    #pragma unroll
    for (int i = 0; i < 4; ++i)
        tile[threadIdx.y + i * 8][threadIdx.x] =
            src[(size_t)(bk + threadIdx.y + i * 8) * N + bn + threadIdx.x];
    __syncthreads();
    #pragma unroll
    for (int i = 0; i < 4; ++i)
        dst[(size_t)(bn + threadIdx.y + i * 8) * K + bk + threadIdx.x] =
            (_Float16)tile[threadIdx.x][threadIdx.y + i * 8];
}

// --------------------------- WMMA tiled GEMM (TDM-fed) ---------------------
// C[M,N] = A[M,K](f16 row-major) * B[N,K](f16, PRE-TRANSPOSED) + bias
// MODE 0: scatter into Q/K/Vt head-major f16 (Q scaled 1/8, V transposed [b,h,d,t])
// MODE 1: write fp32 to Cout
template <int MODE>
__global__ __launch_bounds__(256, 2)
void gemm_wmma(const _Float16* __restrict__ A, const _Float16* __restrict__ BT,
               const float* __restrict__ bias, float* __restrict__ Cout,
               _Float16* __restrict__ qh, _Float16* __restrict__ kh,
               _Float16* __restrict__ vt, int M, int N, int K) {
    __shared__ _Float16 ldsA[2][128][40];   // [buf][m][k] padded to 40 f16 (20 dw)
    __shared__ _Float16 ldsB[2][128][40];   // [buf][n][k]

    const int tid  = threadIdx.x;
    const int lane = tid & 31;
    const int wave = tid >> 5;
    const int nl   = lane & 15;
    const int g    = lane >> 4;
    const int wr   = wave & 3;          // 4 x 32-row groups
    const int wc   = wave >> 2;         // 2 x 64-col groups
    const int bm0  = blockIdx.y * 128;
    const int bn0  = blockIdx.x * 128;

    v8f acc[2][4];
    #pragma unroll
    for (int mi = 0; mi < 2; ++mi)
        #pragma unroll
        for (int ni = 0; ni < 4; ++ni)
            acc[mi][ni] = (v8f){};

    const int ksteps = K >> 5;
    // prologue: DMA first k-tile pair (A 128x32, B 128x32; rows 16dw + 4dw pad)
    if (wave == 0) {
        tdm_load_2d(A  + (size_t)bm0 * K, &ldsA[0][0][0], 32, 128, K, 3, 3);
        tdm_load_2d(BT + (size_t)bn0 * K, &ldsB[0][0][0], 32, 128, K, 3, 3);
    }
    for (int kk = 0; kk < ksteps; ++kk) {
        const int buf = kk & 1;
        if (wave == 0) {
            if (kk + 1 < ksteps) {      // DMA next tile pair into other buffer
                const int k1 = (kk + 1) << 5;
                tdm_load_2d(A  + (size_t)bm0 * K + k1, &ldsA[buf ^ 1][0][0], 32, 128, K, 3, 3);
                tdm_load_2d(BT + (size_t)bn0 * K + k1, &ldsB[buf ^ 1][0][0], 32, 128, K, 3, 3);
                __builtin_amdgcn_s_wait_tensorcnt((short)2);   // current pair done
            } else {
                __builtin_amdgcn_s_wait_tensorcnt((short)0);
            }
        }
        __syncthreads();

        v16h aF[2], bF[4];
        #pragma unroll
        for (int mi = 0; mi < 2; ++mi) {
            const int row = wr * 32 + mi * 16 + nl;
            const int cb  = g * 8;
            ((v8h*)&aF[mi])[0] = *(const v8h*)&ldsA[buf][row][cb];
            ((v8h*)&aF[mi])[1] = *(const v8h*)&ldsA[buf][row][cb + 16];
        }
        #pragma unroll
        for (int ni = 0; ni < 4; ++ni) {
            const int n  = wc * 64 + ni * 16 + nl;
            const int kb = g * 8;
            ((v8h*)&bF[ni])[0] = *(const v8h*)&ldsB[buf][n][kb];
            ((v8h*)&bF[ni])[1] = *(const v8h*)&ldsB[buf][n][kb + 16];
        }
        #pragma unroll
        for (int mi = 0; mi < 2; ++mi)
            #pragma unroll
            for (int ni = 0; ni < 4; ++ni)
                acc[mi][ni] = wmma_f16(aF[mi], bF[ni], acc[mi][ni]);
        __syncthreads();
    }

    // ---- epilogue --------------------------------------------------------
    #pragma unroll
    for (int mi = 0; mi < 2; ++mi) {
        #pragma unroll
        for (int ni = 0; ni < 4; ++ni) {
            #pragma unroll
            for (int r = 0; r < 8; ++r) {
                const int grow = bm0 + wr * 32 + mi * 16 + r + 8 * g;
                const int gcol = bn0 + wc * 64 + ni * 16 + nl;
                float v = acc[mi][ni][r] + bias[gcol];
                if (MODE == 0) {
                    const int sec = gcol >> 10;         // 0:q 1:k 2:v
                    const int c   = gcol & (CB - 1);
                    const int hh  = c >> 6;
                    const int dd  = c & 63;
                    const int bb  = grow >> 11;
                    const int tt  = grow & (TT_ - 1);
                    if (sec == 0) {
                        qh[((((size_t)bb * HH_) + hh) * TT_ + tt) * DD_ + dd] =
                            (_Float16)(v * 0.125f);     // fold 1/sqrt(64)
                    } else if (sec == 1) {
                        kh[((((size_t)bb * HH_) + hh) * TT_ + tt) * DD_ + dd] = (_Float16)v;
                    } else {                            // V transposed: [b,h,d,t]
                        vt[((((size_t)bb * HH_) + hh) * DD_ + dd) * TT_ + tt] = (_Float16)v;
                    }
                } else {
                    Cout[(size_t)grow * N + gcol] = v;
                }
            }
        }
    }
}

// --------------------------- flash attention (TDM-fed) ---------------------
// grid: B*H*(T/128) blocks, 256 threads (8 waves), wave owns 16 query rows.
__global__ __launch_bounds__(256, 2)
void flash_attn(const _Float16* __restrict__ qh, const _Float16* __restrict__ kh,
                const _Float16* __restrict__ vt, _Float16* __restrict__ ah) {
    __shared__ _Float16 ldsK[2][64][72];     // [buf][key][d]   rows 32dw + 4dw pad
    __shared__ _Float16 ldsVt[2][64][72];    // [buf][d][key]
    __shared__ _Float16 pstage[8][16][72];   // per-wave P re-layout bounce

    const int tid  = threadIdx.x;
    const int lane = tid & 31;
    const int wave = tid >> 5;
    const int nl   = lane & 15;
    const int g    = lane >> 4;

    const int qt = blockIdx.x & 15;          // query tile (of 128 rows)
    const int bh = blockIdx.x >> 4;
    const int h  = bh & (HH_ - 1);
    const int b  = bh >> 4;

    const size_t headBase = ((size_t)b * HH_ + h) * TT_;           // [B,H,T,*]
    const _Float16* vtHead = vt + (((size_t)b * HH_ + h) * DD_) * TT_;  // [B,H,D,T]
    const int qbase = qt * 128 + wave * 16;

    // ---- resident Q fragments (16 x 64 f16 = 2 x v16h) -------------------
    v16h qf[2];
    {
        const _Float16* qp = qh + (headBase + qbase + nl) * DD_;
        #pragma unroll
        for (int f = 0; f < 2; ++f) {
            const int db = f * 32 + g * 8;
            ((v8h*)&qf[f])[0] = *(const v8h*)(qp + db);
            ((v8h*)&qf[f])[1] = *(const v8h*)(qp + db + 16);
        }
    }

    float runmax[8], runsum[8];
    v8f   oacc[4];
    #pragma unroll
    for (int r = 0; r < 8; ++r) { runmax[r] = -3.0e38f; runsum[r] = 0.0f; }
    #pragma unroll
    for (int ni = 0; ni < 4; ++ni) oacc[ni] = (v8f){};

    const int nchunks = qt * 2 + 2;          // causal: keys up to end of q tile
    // prologue DMA: K chunk (64x64, stride D) and Vt chunk (64x64, stride T)
    if (wave == 0) {
        tdm_load_2d(kh + (headBase + 0) * DD_, &ldsK[0][0][0],  64, 64, DD_, 4, 3);
        tdm_load_2d(vtHead + 0,                &ldsVt[0][0][0], 64, 64, TT_, 4, 3);
    }
    for (int jc = 0; jc < nchunks; ++jc) {
        const int buf   = jc & 1;
        const int jbase = jc * 64;
        if (wave == 0) {
            if (jc + 1 < nchunks) {
                const int jn = (jc + 1) * 64;
                tdm_load_2d(kh + (headBase + jn) * DD_, &ldsK[buf ^ 1][0][0],  64, 64, DD_, 4, 3);
                tdm_load_2d(vtHead + jn,                &ldsVt[buf ^ 1][0][0], 64, 64, TT_, 4, 3);
                __builtin_amdgcn_s_wait_tensorcnt((short)2);
            } else {
                __builtin_amdgcn_s_wait_tensorcnt((short)0);
            }
        }
        __syncthreads();

        // ---- S = Q K^T  (16x64 per wave, fp32 accum) ---------------------
        v8f sacc[4];
        #pragma unroll
        for (int ni = 0; ni < 4; ++ni) sacc[ni] = (v8f){};
        #pragma unroll
        for (int kc = 0; kc < 2; ++kc) {
            #pragma unroll
            for (int ni = 0; ni < 4; ++ni) {
                v16h bK;
                const int row = ni * 16 + nl;          // key within chunk
                const int kb  = kc * 32 + g * 8;       // d-dim sub-chunk
                ((v8h*)&bK)[0] = *(const v8h*)&ldsK[buf][row][kb];
                ((v8h*)&bK)[1] = *(const v8h*)&ldsK[buf][row][kb + 16];
                sacc[ni] = wmma_f16(qf[kc], bK, sacc[ni]);
            }
        }

        // ---- causal mask + online softmax --------------------------------
        #pragma unroll
        for (int r = 0; r < 8; ++r) {
            const int qrow = qbase + r + 8 * g;
            float mymax = -3.0e38f;
            #pragma unroll
            for (int ni = 0; ni < 4; ++ni) {
                const int key = jbase + ni * 16 + nl;
                if (key > qrow) sacc[ni][r] = -3.0e38f;
                mymax = fmaxf(mymax, sacc[ni][r]);
            }
            #pragma unroll
            for (int m = 1; m < 16; m <<= 1)           // stays within 16-lane half
                mymax = fmaxf(mymax, __shfl_xor(mymax, m, 32));
            const float nm   = fmaxf(runmax[r], mymax);
            const float corr = __expf(runmax[r] - nm);
            runmax[r] = nm;
            float ps = 0.0f;
            #pragma unroll
            for (int ni = 0; ni < 4; ++ni) {
                const float p = __expf(sacc[ni][r] - nm);
                sacc[ni][r] = p;
                ps += p;
            }
            #pragma unroll
            for (int m = 1; m < 16; m <<= 1)
                ps += __shfl_xor(ps, m, 32);
            runsum[r] = runsum[r] * corr + ps;
            #pragma unroll
            for (int ni = 0; ni < 4; ++ni) oacc[ni][r] *= corr;
        }

        // ---- re-layout P (C-layout -> A-layout) through per-wave LDS -----
        #pragma unroll
        for (int r = 0; r < 8; ++r)
            #pragma unroll
            for (int ni = 0; ni < 4; ++ni)
                pstage[wave][r + 8 * g][ni * 16 + nl] = (_Float16)sacc[ni][r];
        asm volatile("s_wait_dscnt 0x0" ::: "memory");   // wave-local LDS RAW

        v16h pf[2];
        #pragma unroll
        for (int kc = 0; kc < 2; ++kc) {
            const int kb = kc * 32 + g * 8;
            ((v8h*)&pf[kc])[0] = *(const v8h*)&pstage[wave][nl][kb];
            ((v8h*)&pf[kc])[1] = *(const v8h*)&pstage[wave][nl][kb + 16];
        }

        // ---- O += P V ----------------------------------------------------
        #pragma unroll
        for (int kc = 0; kc < 2; ++kc) {
            #pragma unroll
            for (int ni = 0; ni < 4; ++ni) {
                v16h bV;
                const int drow = ni * 16 + nl;         // output d dim
                const int kb   = kc * 32 + g * 8;      // key sub-chunk
                ((v8h*)&bV)[0] = *(const v8h*)&ldsVt[buf][drow][kb];
                ((v8h*)&bV)[1] = *(const v8h*)&ldsVt[buf][drow][kb + 16];
                oacc[ni] = wmma_f16(pf[kc], bV, oacc[ni]);
            }
        }
        __syncthreads();
    }

    // ---- normalize and write [B,T,H*D] f16 -------------------------------
    #pragma unroll
    for (int r = 0; r < 8; ++r) {
        const float inv = 1.0f / runsum[r];
        const int qrow  = qbase + r + 8 * g;
        #pragma unroll
        for (int ni = 0; ni < 4; ++ni) {
            const int d = ni * 16 + nl;
            ah[((size_t)b * TT_ + qrow) * CB + h * DD_ + d] =
                (_Float16)(oacc[ni][r] * inv);
        }
    }
}

// --------------------------- host-side launcher ----------------------------
extern "C" void kernel_launch(void* const* d_in, const int* in_sizes, int n_in,
                              void* d_out, int out_size, void* d_ws, size_t ws_size,
                              hipStream_t stream) {
    const float* x     = (const float*)d_in[0];
    const float* w_qkv = (const float*)d_in[1];
    const float* b_qkv = (const float*)d_in[2];
    const float* w_out = (const float*)d_in[3];
    const float* b_out = (const float*)d_in[4];
    float* out = (float*)d_out;

    const int M  = BB_ * TT_;       // 4096
    const int K  = CB;              // 1024
    const int N1 = 3 * CB;          // 3072
    const int N2 = CB;              // 1024

    _Float16* ws = (_Float16*)d_ws;
    size_t off = 0;
    _Float16* xh     = ws + off; off += (size_t)M * K;
    _Float16* wqkvT  = ws + off; off += (size_t)K * N1;   // [N1][K]
    _Float16* woutT  = ws + off; off += (size_t)K * N2;   // [N2][K]
    _Float16* qh     = ws + off; off += (size_t)BB_ * HH_ * TT_ * DD_;
    _Float16* kh     = ws + off; off += (size_t)BB_ * HH_ * TT_ * DD_;
    _Float16* vt     = ws + off; off += (size_t)BB_ * HH_ * TT_ * DD_;  // [b,h,d,t]
    _Float16* ah     = ws + off; off += (size_t)M * CB;

    // 1) convert inputs to f16 (weights transposed to [N][K])
    cvt_f32_f16<<<(M * K) / 1024, 256, 0, stream>>>(x, xh, M * K);
    transpose_cvt<<<dim3(N1 / 32, K / 32), dim3(32, 8), 0, stream>>>(w_qkv, wqkvT, K, N1);
    transpose_cvt<<<dim3(N2 / 32, K / 32), dim3(32, 8), 0, stream>>>(w_out, woutT, K, N2);

    // 2) QKV projection -> head-major q/k (q pre-scaled), v transposed
    gemm_wmma<0><<<dim3(N1 / 128, M / 128), 256, 0, stream>>>(
        xh, wqkvT, b_qkv, nullptr, qh, kh, vt, M, N1, K);

    // 3) causal flash attention
    flash_attn<<<dim3(BB_ * HH_ * (TT_ / 128)), 256, 0, stream>>>(qh, kh, vt, ah);

    // 4) output projection (fp32 result)
    gemm_wmma<1><<<dim3(N2 / 128, M / 128), 256, 0, stream>>>(
        ah, woutT, b_out, out, nullptr, nullptr, nullptr, M, N2, K);
}
